// CharLSTM_11330123727243
// MI455X (gfx1250) — matrix operand order repaired and checked by reading, not compiled, not run on hardware
//
#include <hip/hip_runtime.h>
#include <hip/hip_bf16.h>
#include <cstdint>

// CharLSTM on gfx1250: bf16 WMMA recurrence, weights L2-resident, async-to-LDS
// double-buffered weight streaming, fp32 accumulation.
// B=128, T=256, V=256, D=1024, L=4.

typedef __attribute__((ext_vector_type(16))) __bf16 v16bf;
typedef __attribute__((ext_vector_type(8)))  __bf16 v8bf;
typedef __attribute__((ext_vector_type(8)))  float  v8f;

typedef const __attribute__((address_space(1))) v8bf* gv8p;   // force global_load

#define LSTM_B 128
#define LSTM_T 256
#define LSTM_V 256
#define LSTM_D 1024
#define LSTM_FD 4096   // 4*D
#define LSTM_L 4

// ---------------------------------------------------------------------------
// One-time prep kernels
// ---------------------------------------------------------------------------

// embed [V,D] f32 -> bf16
__global__ void cvt_embed_bf16(const float* __restrict__ embed,
                               __bf16* __restrict__ embed_bf) {
    int tid = blockIdx.x * blockDim.x + threadIdx.x;   // V*D = 262144
    embed_bf[tid] = (__bf16)embed[tid];
}

// Pack [Wx_l; Wh_l] (K=2048, N=4096, row-major f32) into per-stage-contiguous
// WMMA B-fragment layout, bf16:
//   Wp[((((l*64 + ntg)*64 + kt)*4 + f)*32 + lane)*16 + e]
// where nt = ntg*4 + f is the 16-wide N tile, and per fragment:
//   lane<16: K = kt*32 + 0..15 ; lane>=16: K = kt*32 + 16..31 ; N = nt*16+lane%16
// => for fixed (l, ntg, kt) the 4 fragments form one contiguous 4KB block.
__global__ void pack_weights(const float* __restrict__ Wx,
                             const float* __restrict__ Wh,
                             __bf16* __restrict__ Wp) {
    int tid = blockIdx.x * blockDim.x + threadIdx.x;   // 2^21 threads
    int lane = tid & 31;
    int f    = (tid >> 5) & 3;
    int kt   = (tid >> 7) & 63;
    int ntg  = (tid >> 13) & 63;
    int l    = tid >> 19;
    int half = lane >> 4;
    int lrow = lane & 15;
    int n  = (ntg * 4 + f) * 16 + lrow;
    int kb = kt * 32 + half * 16;
    __bf16* dst = Wp + (size_t)tid * 16;
#pragma unroll
    for (int e = 0; e < 16; ++e) {
        int k = kb + e;
        float v = (k < LSTM_D)
            ? Wx[((size_t)l * LSTM_D + k) * LSTM_FD + n]
            : Wh[((size_t)l * LSTM_D + (k - LSTM_D)) * LSTM_FD + n];
        dst[e] = (__bf16)v;
    }
}

// Pack Wout [D=1024, V=256] into B-fragment layout: Wop[((nt*32+kt)*32+lane)][16]
__global__ void pack_wout(const float* __restrict__ Wout,
                          __bf16* __restrict__ Wop) {
    int tid = blockIdx.x * blockDim.x + threadIdx.x;   // 2^14 threads
    int lane = tid & 31;
    int kt   = (tid >> 5) & 31;
    int nt   = tid >> 10;
    int half = lane >> 4;
    int lrow = lane & 15;
    int n  = nt * 16 + lrow;
    int kb = kt * 32 + half * 16;
    __bf16* dst = Wop + (size_t)tid * 16;
#pragma unroll
    for (int e = 0; e < 16; ++e)
        dst[e] = (__bf16)Wout[(size_t)(kb + e) * LSTM_V + n];
}

// zero c (f32) and h (bf16) state: L*B*D each
__global__ void init_state(float* __restrict__ c, __bf16* __restrict__ hbf) {
    int tid = blockIdx.x * blockDim.x + threadIdx.x;   // 524288
    c[tid]   = 0.0f;
    hbf[tid] = (__bf16)0.0f;
}

// ---------------------------------------------------------------------------
// Fragment helpers
// ---------------------------------------------------------------------------
__device__ __forceinline__ v16bf make_frag(const __bf16* p0, const __bf16* p1) {
    v8bf lo = *(const v8bf*)p0;
    v8bf hi = *(const v8bf*)p1;
    return __builtin_shufflevector(lo, hi, 0, 1, 2, 3, 4, 5, 6, 7,
                                           8, 9, 10, 11, 12, 13, 14, 15);
}

// global-address-space A fragment load: guarantees global_load_b128
__device__ __forceinline__ v16bf make_frag_g(const __bf16* p0, const __bf16* p1) {
    v8bf lo = *(gv8p)p0;
    v8bf hi = *(gv8p)p1;
    return __builtin_shufflevector(lo, hi, 0, 1, 2, 3, 4, 5, 6, 7,
                                           8, 9, 10, 11, 12, 13, 14, 15);
}

__device__ __forceinline__ v8f wmma_bf16(v16bf a, v16bf b, v8f c) {
    return __builtin_amdgcn_wmma_f32_16x16x32_bf16(
        false, a, false, b, (short)0, c, false, false);
}

// ---------------------------------------------------------------------------
// Recurrent GEMM: Z[128,4096] = [x_or_h_{l-1} ; h_prev_t] @ [Wx;Wh]
// Block = 4 waves (128 thr), covers all M=128 x 64 columns (4 n-tiles).
// Each wave: 2 m-tiles x 4 n-tiles -> 8 WMMA per k-step, A loaded once.
// B fragments stream Global->LDS via async copies (double buffered),
// shared by all 4 waves. K loop split in halves so A stays pure global.
// grid = 64 blocks.
// ---------------------------------------------------------------------------
__global__ void __launch_bounds__(128)
lstm_gemm(const __bf16* __restrict__ A0,      // h of layer l-1 (null if l==0)
          const int*    __restrict__ gidx,    // idx + t (null if l>0)
          int gstride,                        // = T
          const __bf16* __restrict__ embed_bf,
          const __bf16* __restrict__ Ah,      // h_prev of this layer
          const __bf16* __restrict__ Wl,      // packed [Wx_l;Wh_l]
          float* __restrict__ Z) {
    __shared__ __align__(16) __bf16 bstage[2][2048];   // 2 x 4KB stages

    const int tid  = threadIdx.x;
    const int lane = tid & 31;
    const int wid  = tid >> 5;        // 0..3  -> m-pair
    const int half = lane >> 4;
    const int lrow = lane & 15;
    const int ntg  = blockIdx.x;      // 0..63 -> 4 n-tiles

    // A row pointers for this wave's two m-tiles
    const __bf16* arow0[2];
    const __bf16* arow1[2];
#pragma unroll
    for (int mi = 0; mi < 2; ++mi) {
        int row = (wid * 2 + mi) * 16 + lrow;
        if (gidx) arow0[mi] = embed_bf + (size_t)gidx[(size_t)row * gstride] * LSTM_D;
        else      arow0[mi] = A0 + (size_t)row * LSTM_D;
        arow1[mi] = Ah + (size_t)row * LSTM_D;
    }

    // B stream for this block: 64 k-stages x 4KB, contiguous per stage
    const __bf16* gsrc = Wl + (size_t)ntg * (64 * 2048);

    // async copy of one 4KB stage: each thread moves 32B (2 x b128)
    auto issue = [&](int kt, int st) {
        const __bf16* g = gsrc + (size_t)kt * 2048 + tid * 16;
        unsigned lds = (unsigned)(uintptr_t)&bstage[st][tid * 16];
        asm volatile(
            "global_load_async_to_lds_b128 %0, %1, off\n\t"
            "global_load_async_to_lds_b128 %0, %1, off offset:16"
            :: "v"(lds), "v"(g) : "memory");
    };

    v8f acc[2][4] = {};

    // one k-step: consume stage kt, keep stage kt+1 in flight
    auto kstep = [&](int kt, const __bf16* apA, const __bf16* apB) {
        const int st = kt & 1;
        if (kt < 63) {
            issue(kt + 1, st ^ 1);
            // 2 older async ops must retire (in-order) -> stage kt resident
            asm volatile("s_wait_asynccnt 0x2" ::: "memory");
        } else {
            asm volatile("s_wait_asynccnt 0x0" ::: "memory");
        }
        __syncthreads();                           // stage st visible to all waves

        v16bf a0 = make_frag_g(apA + half * 16, apA + half * 16 + 8);
        v16bf a1 = make_frag_g(apB + half * 16, apB + half * 16 + 8);

        // load all 4 B fragments first -> clause the ds_loads, single wait
        v16bf bm[4];
#pragma unroll
        for (int f = 0; f < 4; ++f) {
            const __bf16* bp = &bstage[st][f * 512 + lane * 16];
            bm[f] = make_frag(bp, bp + 8);
        }
#pragma unroll
        for (int f = 0; f < 4; ++f) {
            acc[0][f] = wmma_bf16(a0, bm[f], acc[0][f]);
            acc[1][f] = wmma_bf16(a1, bm[f], acc[1][f]);
        }
        __syncthreads();                           // done reading stage st
    };

    issue(0, 0);                                   // prologue: stage 0
#pragma unroll 1
    for (int kt = 0; kt < 32; ++kt)                // x / h_{l-1} half
        kstep(kt, arow0[0] + kt * 32, arow0[1] + kt * 32);
#pragma unroll 1
    for (int kt = 32; kt < 64; ++kt)               // recurrent h half
        kstep(kt, arow1[0] + (kt - 32) * 32, arow1[1] + (kt - 32) * 32);

#pragma unroll
    for (int mi = 0; mi < 2; ++mi)
#pragma unroll
        for (int f = 0; f < 4; ++f)
#pragma unroll
            for (int j = 0; j < 8; ++j)
                Z[(size_t)((wid * 2 + mi) * 16 + j + half * 8) * LSTM_FD
                  + (ntg * 4 + f) * 16 + lrow] = acc[mi][f][j];
}

// ---------------------------------------------------------------------------
// Gate pointwise: c,h update. One thread per (b,d) = 131072 threads.
// ---------------------------------------------------------------------------
__device__ __forceinline__ float sigf(float x) { return 1.0f / (1.0f + __expf(-x)); }

__global__ void lstm_gate(const float* __restrict__ Z,
                          const float* __restrict__ bias,   // b + l*4096
                          float* __restrict__ c,            // c state of layer l
                          __bf16* __restrict__ hbf) {       // h state of layer l
    int tid = blockIdx.x * blockDim.x + threadIdx.x;
    int b   = tid >> 10;
    int d   = tid & 1023;
    const float* z = Z + (size_t)b * LSTM_FD;
    float zi = z[d]            + bias[d];
    float zf = z[LSTM_D + d]   + bias[LSTM_D + d];
    float zg = z[2*LSTM_D + d] + bias[2*LSTM_D + d];
    float zo = z[3*LSTM_D + d] + bias[3*LSTM_D + d];
    float cn = sigf(zf) * c[tid] + sigf(zi) * tanhf(zg);
    float h  = sigf(zo) * tanhf(cn);
    c[tid]   = cn;
    hbf[tid] = (__bf16)h;
}

// ---------------------------------------------------------------------------
// Output projection for one timestep: logits[:, t, :] = h3 @ Wout
// M=128 x N=256, K=1024 -> 128 tiles, 32 blocks x 4 waves, wave per tile.
// ---------------------------------------------------------------------------
__global__ void proj_gemm(const __bf16* __restrict__ H3,   // [128,1024] bf16
                          const __bf16* __restrict__ Wop,  // packed Wout
                          float* __restrict__ out,         // [B,T,V]
                          int t) {
    int lane = threadIdx.x & 31;
    int w    = blockIdx.x * 4 + (threadIdx.x >> 5);
    int mt   = w & 7;          // 8 M tiles
    int nt   = w >> 3;         // 16 N tiles
    int half = lane >> 4;
    int lrow = lane & 15;
    int row  = mt * 16 + lrow;

    const __bf16* arow  = H3 + (size_t)row * LSTM_D;
    const __bf16* bbase = Wop + (((size_t)nt * 32) * 32 + lane) * 16;

    v8f acc = {};
#pragma unroll 4
    for (int kt = 0; kt < 32; ++kt) {
        int k0 = half * 16;
        v16bf a = make_frag_g(arow + kt * 32 + k0, arow + kt * 32 + k0 + 8);
        const __bf16* bp = bbase + (size_t)kt * (32 * 16);
        __builtin_prefetch((const void*)(bp + 2 * 32 * 16), 0, 1);
        v16bf bm = make_frag_g(bp, bp + 8);
        acc = wmma_bf16(a, bm, acc);
    }

#pragma unroll
    for (int j = 0; j < 8; ++j) {
        int b = mt * 16 + j + half * 8;
        out[((size_t)b * LSTM_T + t) * LSTM_V + nt * 16 + lrow] = acc[j];
    }
}

// ---------------------------------------------------------------------------
// Host-side orchestration
// ---------------------------------------------------------------------------
extern "C" void kernel_launch(void* const* d_in, const int* in_sizes, int n_in,
                              void* d_out, int out_size, void* d_ws, size_t ws_size,
                              hipStream_t stream) {
    (void)in_sizes; (void)n_in; (void)out_size; (void)ws_size;
    const int*   idx   = (const int*)d_in[0];    // [B,T]
    const float* embed = (const float*)d_in[1];  // [V,D]
    const float* Wx    = (const float*)d_in[2];  // [L,D,4D]
    const float* Wh    = (const float*)d_in[3];  // [L,D,4D]
    const float* bias  = (const float*)d_in[4];  // [L,4D]
    const float* Wout  = (const float*)d_in[5];  // [D,V]
    float* out = (float*)d_out;                  // [B,T,V]

    char* ws = (char*)d_ws;
    size_t off = 0;
    auto alloc = [&](size_t bytes) {
        size_t o = off;
        off = (off + bytes + 255) & ~(size_t)255;
        return o;
    };
    const size_t BD = (size_t)LSTM_B * LSTM_D;           // 131072
    __bf16* embed_bf = (__bf16*)(ws + alloc((size_t)LSTM_V * LSTM_D * 2));
    __bf16* Wp       = (__bf16*)(ws + alloc((size_t)LSTM_L * 64 * 64 * 2048 * 2)); // 64MB
    __bf16* Wop      = (__bf16*)(ws + alloc((size_t)16 * 32 * 32 * 16 * 2));
    float*  cst      = (float*) (ws + alloc((size_t)LSTM_L * BD * 4));
    __bf16* hbf      = (__bf16*)(ws + alloc((size_t)LSTM_L * BD * 2));
    float*  Z        = (float*) (ws + alloc((size_t)LSTM_B * LSTM_FD * 4));

    // one-time prep (re-run every call for determinism)
    cvt_embed_bf16<<<1024, 256, 0, stream>>>(embed, embed_bf);
    pack_weights<<<8192, 256, 0, stream>>>(Wx, Wh, Wp);
    pack_wout<<<64, 256, 0, stream>>>(Wout, Wop);
    init_state<<<2048, 256, 0, stream>>>(cst, hbf);

    const size_t WL = (size_t)64 * 64 * 2048;   // packed elems per layer
    for (int t = 0; t < LSTM_T; ++t) {
        for (int l = 0; l < LSTM_L; ++l) {
            const __bf16* A0   = (l == 0) ? nullptr : hbf + (size_t)(l - 1) * BD;
            const int*    gidx = (l == 0) ? (idx + t) : nullptr;
            lstm_gemm<<<64, 128, 0, stream>>>(A0, gidx, LSTM_T, embed_bf,
                                              hbf + (size_t)l * BD,
                                              Wp + (size_t)l * WL, Z);
            lstm_gate<<<512, 256, 0, stream>>>(Z, bias + (size_t)l * LSTM_FD,
                                               cst + (size_t)l * BD,
                                               hbf + (size_t)l * BD);
        }
        proj_gemm<<<32, 128, 0, stream>>>(hbf + (size_t)3 * BD, Wop, out, t);
    }
}
